// TopkRouter_38663295599096
// MI455X (gfx1250) — matrix operand order repaired and checked by reading, not compiled, not run on hardware
//
#include <hip/hip_runtime.h>
#include <hip/hip_bf16.h>

// ---------------- Problem constants (match reference) ----------------
#define N_EXPERTS  64
#define N_GROUP    8
#define GSIZE      8          // experts per group
#define TOPK_GROUP 4
#define TOP_K      8
#define HIDDEN     2048
#define N_TOKENS   32768
#define ROUTER_SCALING 1.0f

// ---------------- Tiling ----------------
#define BLOCK      256        // 8 wave32
#define TOK_PER_WG 128        // 8 waves x 16 tokens
#define LDSW       2056       // padded LDS row stride (bf16 elems) = 2048 + 8

typedef __attribute__((ext_vector_type(16))) __bf16 v16bf;
typedef __attribute__((ext_vector_type(8)))  __bf16 v8bf;
typedef __attribute__((ext_vector_type(4)))  __bf16 v4bf;
typedef __attribute__((ext_vector_type(8)))  float  v8f;

#define NEG_INF (-__builtin_inff())

// LDS byte offsets
#define LDS_W_BYTES   (N_EXPERTS * LDSW * 2)                 // 263168
#define LDS_LG_BYTES  (TOK_PER_WG * N_EXPERTS * 4)           // 32768
#define LDS_TOTAL     (LDS_W_BYTES + LDS_LG_BYTES + N_EXPERTS * 4)

// __launch_bounds__(threads, minWavesPerEU): LDS (289 KB) already limits us to
// 1 WG/WGP == 2 waves/SIMD, so tell the compiler 2 — relaxes the VGPR budget
// enough to keep all four B tiles live simultaneously.
__global__ __launch_bounds__(BLOCK, 2)
void router_topk_kernel(const float* __restrict__ hs,      // [N_TOKENS, HIDDEN]
                        const float* __restrict__ wgt,     // [N_EXPERTS, HIDDEN]
                        const float* __restrict__ bias,    // [N_EXPERTS]
                        float* __restrict__ out_w,         // [N_TOKENS, TOP_K]
                        int*   __restrict__ out_i)         // [N_TOKENS, TOP_K]
{
    extern __shared__ char smem[];
    __bf16* lds_w    = (__bf16*)smem;                            // [64][LDSW] bf16
    float*  lds_lg   = (float*)(smem + LDS_W_BYTES);             // [128][64] f32 logits
    float*  lds_bias = (float*)(smem + LDS_W_BYTES + LDS_LG_BYTES); // [64] f32

    const int tid = threadIdx.x;

    // ---------------- Stage bias + weight (fp32 -> bf16) into LDS ----------------
    if (tid < N_EXPERTS) lds_bias[tid] = bias[tid];

    for (int idx = tid; idx < N_EXPERTS * (HIDDEN / 4); idx += BLOCK) {
        const int row = idx >> 9;          // / (HIDDEN/4) == 512
        const int c4  = idx & 511;
        float4 w = *(const float4*)(wgt + (size_t)row * HIDDEN + c4 * 4);
        v4bf pk;
        pk[0] = (__bf16)w.x; pk[1] = (__bf16)w.y;
        pk[2] = (__bf16)w.z; pk[3] = (__bf16)w.w;
        *(v4bf*)(lds_w + row * LDSW + c4 * 4) = pk;              // ds_store_b64
    }
    __syncthreads();

    // ---------------- WMMA logits: each wave = 16 tokens x 64 experts ----------------
    const int lane = tid & 31;
    const int wave = tid >> 5;
    const int hl   = lane >> 4;       // half-wave: 0 or 1
    const int mn   = lane & 15;       // M row (A) / N col (B)
    const int tokBase = blockIdx.x * TOK_PER_WG + wave * 16;

    const float*  arow  = hs + (size_t)(tokBase + mn) * HIDDEN + hl * 16;
    // Per-expert-tile B row bases (constant across K loop; K becomes an index offset)
    const __bf16* brow0 = lds_w + (0 * 16 + mn) * LDSW + hl * 16;
    const __bf16* brow1 = lds_w + (1 * 16 + mn) * LDSW + hl * 16;
    const __bf16* brow2 = lds_w + (2 * 16 + mn) * LDSW + hl * 16;
    const __bf16* brow3 = lds_w + (3 * 16 + mn) * LDSW + hl * 16;

    v8f acc0 = {}, acc1 = {}, acc2 = {}, acc3 = {};

#pragma unroll 4
    for (int k0 = 0; k0 < HIDDEN; k0 += 32) {
        // speculative L2 prefetch of upcoming A data (global_prefetch_b8)
        __builtin_prefetch(arow + k0 + 256, 0, 1);

        // --- A operand (global, fp32): K contiguous per lane half, same order as B ---
        // lanes 0-15: K = k0+[0..15]; lanes 16-31: K = k0+[16..31]
        float4 a0 = *(const float4*)(arow + k0);
        float4 a1 = *(const float4*)(arow + k0 + 4);
        float4 a2 = *(const float4*)(arow + k0 + 8);
        float4 a3 = *(const float4*)(arow + k0 + 12);

        // --- All four B tiles (8 independent ds_load_b128, one clause) ---
        v8bf b00 = *(const v8bf*)(brow0 + k0);
        v8bf b01 = *(const v8bf*)(brow0 + k0 + 8);
        v8bf b10 = *(const v8bf*)(brow1 + k0);
        v8bf b11 = *(const v8bf*)(brow1 + k0 + 8);
        v8bf b20 = *(const v8bf*)(brow2 + k0);
        v8bf b21 = *(const v8bf*)(brow2 + k0 + 8);
        v8bf b30 = *(const v8bf*)(brow3 + k0);
        v8bf b31 = *(const v8bf*)(brow3 + k0 + 8);

        v16bf A;
        A[0]  = (__bf16)a0.x; A[1]  = (__bf16)a0.y; A[2]  = (__bf16)a0.z; A[3]  = (__bf16)a0.w;
        A[4]  = (__bf16)a1.x; A[5]  = (__bf16)a1.y; A[6]  = (__bf16)a1.z; A[7]  = (__bf16)a1.w;
        A[8]  = (__bf16)a2.x; A[9]  = (__bf16)a2.y; A[10] = (__bf16)a2.z; A[11] = (__bf16)a2.w;
        A[12] = (__bf16)a3.x; A[13] = (__bf16)a3.y; A[14] = (__bf16)a3.z; A[15] = (__bf16)a3.w;

        v16bf B0, B1, B2, B3;
#pragma unroll
        for (int q = 0; q < 8; ++q) {
            B0[q] = b00[q]; B0[8 + q] = b01[q];
            B1[q] = b10[q]; B1[8 + q] = b11[q];
            B2[q] = b20[q]; B2[8 + q] = b21[q];
            B3[q] = b30[q]; B3[8 + q] = b31[q];
        }

        // Scheduling fence: forces all 8 LDS loads + A converts to complete
        // before ANY wmma issues -> one s_wait_dscnt for the whole quartet,
        // four back-to-back v_wmma pipelining in the matrix unit.
        asm volatile("" : "+v"(A), "+v"(B0), "+v"(B1), "+v"(B2), "+v"(B3));

        acc0 = __builtin_amdgcn_wmma_f32_16x16x32_bf16(false, A, false, B0, (short)0, acc0, false, false);
        acc1 = __builtin_amdgcn_wmma_f32_16x16x32_bf16(false, A, false, B1, (short)0, acc1, false, false);
        acc2 = __builtin_amdgcn_wmma_f32_16x16x32_bf16(false, A, false, B2, (short)0, acc2, false, false);
        acc3 = __builtin_amdgcn_wmma_f32_16x16x32_bf16(false, A, false, B3, (short)0, acc3, false, false);
    }

    // ---------------- Scatter accumulators to LDS logits [tokLocal][expert] ----------------
    // C layout: VGPR r -> lanes 0-15: M=r, lanes 16-31: M=8+r; N = lane&15 (+16j)
    {
        float* lgbase = lds_lg + (wave * 16 + hl * 8) * N_EXPERTS + mn;
#pragma unroll
        for (int r = 0; r < 8; ++r) {
            lgbase[r * N_EXPERTS + 0]  = acc0[r];
            lgbase[r * N_EXPERTS + 16] = acc1[r];
            lgbase[r * N_EXPERTS + 32] = acc2[r];
            lgbase[r * N_EXPERTS + 48] = acc3[r];
        }
    }
    __syncthreads();

    // ---------------- Fused grouped top-k: one thread per token ----------------
    if (tid < TOK_PER_WG) {
        float* lg = lds_lg + tid * N_EXPERTS;

        // sigmoid + bias, in place (sfc); recover sigmoid later as sfc - bias
#pragma unroll 8
        for (int e = 0; e < N_EXPERTS; ++e) {
            float x  = lg[e];
            float sc = 1.0f / (1.0f + __expf(-x));
            lg[e] = sc + lds_bias[e];
        }

        // group scores: sum of top-4 of 8 within each group
        float gs[N_GROUP];
#pragma unroll
        for (int g = 0; g < N_GROUP; ++g) {
            float v[GSIZE];
#pragma unroll
            for (int j = 0; j < GSIZE; ++j) v[j] = lg[g * GSIZE + j];
            float ssum = 0.0f;
#pragma unroll
            for (int it = 0; it < TOPK_GROUP; ++it) {
                float mx = v[0]; int mi = 0;
#pragma unroll
                for (int j = 1; j < GSIZE; ++j) { if (v[j] > mx) { mx = v[j]; mi = j; } }
                ssum += mx;
#pragma unroll
                for (int j = 0; j < GSIZE; ++j) v[j] = (j == mi) ? NEG_INF : v[j];
            }
            gs[g] = ssum;
        }

        // top-4 groups -> mask
        unsigned gmask = 0u;
#pragma unroll
        for (int it = 0; it < TOPK_GROUP; ++it) {
            float mx = gs[0]; int mi = 0;
#pragma unroll
            for (int g = 1; g < N_GROUP; ++g) { if (gs[g] > mx) { mx = gs[g]; mi = g; } }
            gmask |= (1u << mi);
#pragma unroll
            for (int g = 0; g < N_GROUP; ++g) gs[g] = (g == mi) ? NEG_INF : gs[g];
        }

        // top-8 experts among masked sfc
        unsigned long long used = 0ull;
        float tw[TOP_K]; int ti[TOP_K];
        float wsum = 0.0f;
#pragma unroll
        for (int it = 0; it < TOP_K; ++it) {
            float mx = NEG_INF; int mi = 0;
#pragma unroll 4
            for (int e = 0; e < N_EXPERTS; ++e) {
                float sc = lg[e];
                bool ok = (((gmask >> (e >> 3)) & 1u) != 0u) && (((used >> e) & 1ull) == 0ull);
                if (ok && sc > mx) { mx = sc; mi = e; }
            }
            used |= (1ull << mi);
            float s = mx - lds_bias[mi];   // recover differentiable sigmoid score
            tw[it] = s; ti[it] = mi; wsum += s;
        }

        const float inv = ROUTER_SCALING / (wsum + 1e-20f);
        const int tok = blockIdx.x * TOK_PER_WG + tid;
#pragma unroll
        for (int i = 0; i < TOP_K; ++i) {
            out_w[(size_t)tok * TOP_K + i] = tw[i] * inv;
            out_i[(size_t)tok * TOP_K + i] = ti[i];
        }
    }
}

extern "C" void kernel_launch(void* const* d_in, const int* in_sizes, int n_in,
                              void* d_out, int out_size, void* d_ws, size_t ws_size,
                              hipStream_t stream) {
    const float* hs   = (const float*)d_in[0];   // hidden_states [32768, 2048] f32
    const float* wgt  = (const float*)d_in[1];   // weight        [64, 2048]    f32
    const float* bias = (const float*)d_in[2];   // e_score_correction_bias [64] f32

    float* out_w = (float*)d_out;                                    // [32768, 8] f32
    int*   out_i = (int*)((float*)d_out + (size_t)N_TOKENS * TOP_K); // [32768, 8] i32

    const size_t shmem = LDS_TOTAL;  // ~289.3 KB < 320 KB WGP LDS
    (void)hipFuncSetAttribute((const void*)router_topk_kernel,
                              hipFuncAttributeMaxDynamicSharedMemorySize, (int)shmem);

    router_topk_kernel<<<N_TOKENS / TOK_PER_WG, BLOCK, shmem, stream>>>(
        hs, wgt, bias, out_w, out_i);
}
